// fSelfAttention_83313775608371
// MI455X (gfx1250) — compile-verified, matrix operand order checked by reading, not run on hardware
//
#include <hip/hip_runtime.h>
#include <hip/hip_bf16.h>

// ---------------------------------------------------------------------------
// Self-attention block for MI455X (gfx1250), bf16 WMMA with f32 accumulation.
//   1) convert x, w_in, w_out -> bf16 (workspace)
//   2) qkv  = x @ w_in^T            (bf16 NT-GEMM, v_wmma_f32_16x16x32_bf16)
//   3) flash attention per (b,h,q-tile), online softmax, mask-before-scale
//   4) out  = attn @ w_out^T + b    (bf16 NT-GEMM, f32 output)
// LDS staging uses GLOBAL_LOAD_ASYNC_TO_LDS_B128 via inline asm (ASYNCcnt),
// with a 16-byte register-copy fallback off-gfx1250.
// ---------------------------------------------------------------------------

typedef __bf16 bf16x16 __attribute__((ext_vector_type(16)));
typedef float  f32x8   __attribute__((ext_vector_type(8)));

#define BS 4
#define NS 2048
#define DM 1024
#define NH 16
#define DH 64
#define E3 3072  // 3*DM

#if defined(__gfx1250__)
#define HAVE_ASYNC_LDS 1
#else
#define HAVE_ASYNC_LDS 0
#endif

// copy 16 bytes global -> LDS (async DMA on gfx1250, tracked by ASYNCcnt)
__device__ __forceinline__ void cp16(unsigned short* l, const unsigned short* g) {
#if HAVE_ASYNC_LDS
    asm volatile("global_load_async_to_lds_b128 %0, %1, off"
                 :
                 : "v"((unsigned)(unsigned long long)l),
                   "v"((unsigned long long)g)
                 : "memory");
#else
    *(uint4*)l = *(const uint4*)g;
#endif
}
__device__ __forceinline__ void wait_async() {
#if HAVE_ASYNC_LDS
#if defined(__has_builtin)
#if __has_builtin(__builtin_amdgcn_s_wait_asynccnt)
#define USED_WAIT_BUILTIN 1
    __builtin_amdgcn_s_wait_asynccnt(0);
#endif
#endif
#ifndef USED_WAIT_BUILTIN
    asm volatile("s_wait_asynccnt 0x0" ::: "memory");
#endif
#endif
}

__device__ __forceinline__ unsigned short f2bf(float f) {
    union { float f; unsigned u; } v; v.f = f;
    unsigned r = v.u + 0x7FFFu + ((v.u >> 16) & 1u);   // RNE
    return (unsigned short)(r >> 16);
}
__device__ __forceinline__ __bf16 us2bf(unsigned short u) {
    return __builtin_bit_cast(__bf16, u);
}
// CDNA5 16-bit A/B fragment K-index pattern: lane-group g (= lane>>4)
// element i in 0..15 -> k = (i<8 ? 0 : 16) + g*8 + (i&7)
__device__ __forceinline__ int kof(int i, int g) {
    return ((i & 8) << 1) + (g << 3) + (i & 7);
}

// ---------------------------------------------------------------------------
// f32 -> bf16 conversion pass
// ---------------------------------------------------------------------------
__global__ void convert_bf16_kernel(const float* __restrict__ src,
                                    unsigned short* __restrict__ dst, int n) {
    int i = blockIdx.x * blockDim.x + threadIdx.x;
    int stride = gridDim.x * blockDim.x;
    for (; i < n; i += stride) dst[i] = f2bf(src[i]);
}

// ---------------------------------------------------------------------------
// C[M,N] = A[M,K] @ B[N,K]^T   (bf16 in, f32 accumulate)
// block = 128 threads (4 waves), tile 128(M) x 64(N), K step 32.
// Wave w owns rows [w*32, w*32+32): 2 row-blocks x 4 col-fragments
// -> 8 WMMAs per K-step, B fragments reused across both row-blocks.
// ---------------------------------------------------------------------------
template <bool OUT_F32>
__global__ void __launch_bounds__(128, 1)
gemm_nt_kernel(const unsigned short* __restrict__ A,
               const unsigned short* __restrict__ B,
               void* __restrict__ C,
               const float* __restrict__ bias,
               int M, int N, int K) {
    __shared__ unsigned short sA[128 * 32];
    __shared__ unsigned short sB[64 * 32];

    const int tid  = threadIdx.x;
    const int lane = tid & 31;
    const int wv   = tid >> 5;
    const int g    = lane >> 4;
    const int ln   = lane & 15;
    const int bm   = blockIdx.x * 128;
    const int bn   = blockIdx.y * 64;

    f32x8 acc[2][4] = {};

    for (int k0 = 0; k0 < K; k0 += 32) {
        // cooperative stage: A 128x32, B 64x32 (16B units)
        for (int idx = tid; idx < 512; idx += 128) {          // sA: 128 rows x 4
            int r = idx >> 2, c = (idx & 3) << 3;
            cp16(&sA[r * 32 + c], &A[(size_t)(bm + r) * K + k0 + c]);
        }
        for (int idx = tid; idx < 256; idx += 128) {          // sB: 64 rows x 4
            int r = idx >> 2, c = (idx & 3) << 3;
            cp16(&sB[r * 32 + c], &B[(size_t)(bn + r) * K + k0 + c]);
        }
        if (k0 + 32 < K) {  // prefetch next K-step (global_prefetch_b8)
            __builtin_prefetch(&A[(size_t)(bm + tid) * K + k0 + 32], 0, 3);
            if (tid < 64)
                __builtin_prefetch(&B[(size_t)(bn + tid) * K + k0 + 32], 0, 3);
        }
        wait_async();
        __syncthreads();

        bf16x16 af[2];
#pragma unroll
        for (int rb = 0; rb < 2; ++rb)
#pragma unroll
            for (int i = 0; i < 16; ++i)
                af[rb][i] = us2bf(sA[(wv * 32 + rb * 16 + ln) * 32 + kof(i, g)]);
#pragma unroll
        for (int f = 0; f < 4; ++f) {
            bf16x16 bf;
#pragma unroll
            for (int i = 0; i < 16; ++i)
                bf[i] = us2bf(sB[(f * 16 + ln) * 32 + kof(i, g)]);
#pragma unroll
            for (int rb = 0; rb < 2; ++rb)
                acc[rb][f] = __builtin_amdgcn_wmma_f32_16x16x32_bf16(
                    false, af[rb], false, bf, (short)0, acc[rb][f], false, false);
        }
        __syncthreads();
    }

#pragma unroll
    for (int rb = 0; rb < 2; ++rb)
#pragma unroll
        for (int f = 0; f < 4; ++f) {
            int col = bn + f * 16 + ln;
#pragma unroll
            for (int r = 0; r < 8; ++r) {
                int row = bm + wv * 32 + rb * 16 + g * 8 + r;
                if constexpr (OUT_F32) {
                    ((float*)C)[(size_t)row * N + col] = acc[rb][f][r] + bias[col];
                } else {
                    ((unsigned short*)C)[(size_t)row * N + col] = f2bf(acc[rb][f][r]);
                }
            }
        }
}

// ---------------------------------------------------------------------------
// Flash attention: one block per (b, h, q-tile of 64 rows), 128 threads.
// qkv bf16 layout: row = b*NS+n (stride 3072), cols [0,1024)=Q, [1024,2048)=K,
// [2048,3072)=V; head slice = h*64.  Output: [b, n, h, dh] bf16 (== [b,n,d]).
// Mask added BEFORE 1/8 scale (faithful to reference).
// ---------------------------------------------------------------------------
__global__ void __launch_bounds__(128, 1)
attn_kernel(const unsigned short* __restrict__ qkv,
            const int* __restrict__ ids,
            unsigned short* __restrict__ out) {
    __shared__ unsigned short sQ[64 * 64];
    __shared__ unsigned short sK[64 * 64];
    __shared__ unsigned short sV[64 * 64];
    __shared__ float sMask[64];
    __shared__ unsigned short sP[4][16 * 64];  // per-wave P relayout scratch

    const int tid  = threadIdx.x;
    const int lane = tid & 31;
    const int wv   = tid >> 5;
    const int g    = lane >> 4;
    const int ln   = lane & 15;

    const int qt = blockIdx.x & 31;
    const int h  = (blockIdx.x >> 5) & 15;
    const int b  = blockIdx.x >> 9;

    const size_t rowbase = (size_t)b * NS;
    const int qcol = h * DH, kcol = DM + h * DH, vcol = 2 * DM + h * DH;

    // stage Q tile (64 q-rows x 64 dh), 16B units: 64 rows x 8 segments
    for (int idx = tid; idx < 512; idx += 128) {
        int r = idx >> 3, c = (idx & 7) << 3;
        cp16(&sQ[r * 64 + c], &qkv[(rowbase + qt * 64 + r) * E3 + qcol + c]);
    }
    wait_async();
    __syncthreads();

    // Q A-fragments, resident for whole kernel (row = wv*16+ln, dh in 2 chunks)
    bf16x16 aq[2];
#pragma unroll
    for (int c = 0; c < 2; ++c)
#pragma unroll
        for (int i = 0; i < 16; ++i)
            aq[c][i] = us2bf(sQ[(wv * 16 + ln) * 64 + c * 32 + kof(i, g)]);

    f32x8 accO[4] = {};
    float rmax[8], rsum[8];
#pragma unroll
    for (int r = 0; r < 8; ++r) { rmax[r] = -3.0e38f; rsum[r] = 0.f; }

    for (int kt = 0; kt < NS / 64; ++kt) {
        __syncthreads();
        for (int idx = tid; idx < 512; idx += 128) {
            int r = idx >> 3, c = (idx & 7) << 3;
            size_t row = (rowbase + kt * 64 + r) * E3;
            cp16(&sK[r * 64 + c], &qkv[row + kcol + c]);
            cp16(&sV[r * 64 + c], &qkv[row + vcol + c]);
        }
        if (tid < 64)
            sMask[tid] = -1.0e9f * (float)ids[b * NS + kt * 64 + tid];
        if (kt + 1 < NS / 64) {  // prefetch next K/V tile rows
            size_t nrow = (rowbase + (kt + 1) * 64 + (tid >> 1)) * E3;
            __builtin_prefetch(&qkv[nrow + ((tid & 1) ? kcol : vcol)], 0, 3);
        }
        wait_async();
        __syncthreads();

        // S = Q @ K^T : per wave 16(q) x 64(keys), accumulate over dh=64
        f32x8 s[4] = {};
#pragma unroll
        for (int f = 0; f < 4; ++f) {
#pragma unroll
            for (int c = 0; c < 2; ++c) {
                bf16x16 bk;
#pragma unroll
                for (int i = 0; i < 16; ++i)
                    bk[i] = us2bf(sK[(f * 16 + ln) * 64 + c * 32 + kof(i, g)]);
                s[f] = __builtin_amdgcn_wmma_f32_16x16x32_bf16(
                    false, aq[c], false, bk, (short)0, s[f], false, false);
            }
        }

        // mask (pre-scale) then scale
#pragma unroll
        for (int f = 0; f < 4; ++f) {
            float mk = sMask[f * 16 + ln];
#pragma unroll
            for (int r = 0; r < 8; ++r) s[f][r] = (s[f][r] + mk) * 0.125f;
        }

        // online softmax: rows of a C-fragment live across 16 lanes in a group
#pragma unroll
        for (int r = 0; r < 8; ++r) {
            float m = fmaxf(fmaxf(s[0][r], s[1][r]), fmaxf(s[2][r], s[3][r]));
#pragma unroll
            for (int off = 1; off < 16; off <<= 1)
                m = fmaxf(m, __shfl_xor(m, off, 32));
            float nm = fmaxf(rmax[r], m);
            float alpha = __expf(rmax[r] - nm);
            rmax[r] = nm;
            float ps = 0.f;
#pragma unroll
            for (int f = 0; f < 4; ++f) {
                s[f][r] = __expf(s[f][r] - nm);
                ps += s[f][r];
            }
#pragma unroll
            for (int off = 1; off < 16; off <<= 1)
                ps += __shfl_xor(ps, off, 32);
            rsum[r] = rsum[r] * alpha + ps;
#pragma unroll
            for (int f = 0; f < 4; ++f) accO[f][r] *= alpha;
        }

        // relayout P: C-fragment (row across lanes) -> A-fragment (row in lane)
#pragma unroll
        for (int f = 0; f < 4; ++f)
#pragma unroll
            for (int r = 0; r < 8; ++r)
                sP[wv][(g * 8 + r) * 64 + f * 16 + ln] = f2bf(s[f][r]);
        // same-wave DS ops are in-order; no block barrier needed (private scratch)

        bf16x16 ap[2];
#pragma unroll
        for (int c = 0; c < 2; ++c)
#pragma unroll
            for (int i = 0; i < 16; ++i)
                ap[c][i] = us2bf(sP[wv][ln * 64 + c * 32 + kof(i, g)]);

        // O += P @ V : K-dim = keys(64), N = dh(64)
#pragma unroll
        for (int f = 0; f < 4; ++f) {
#pragma unroll
            for (int c = 0; c < 2; ++c) {
                bf16x16 bv;
#pragma unroll
                for (int i = 0; i < 16; ++i)
                    bv[i] = us2bf(sV[(c * 32 + kof(i, g)) * 64 + f * 16 + ln]);
                accO[f] = __builtin_amdgcn_wmma_f32_16x16x32_bf16(
                    false, ap[c], false, bv, (short)0, accO[f], false, false);
            }
        }
    }

    // normalize and store to [b, n, h, dh] (== [b, n, d] with d = h*64+dh)
#pragma unroll
    for (int f = 0; f < 4; ++f) {
        int dcol = h * DH + f * 16 + ln;
#pragma unroll
        for (int r = 0; r < 8; ++r) {
            int qrow = qt * 64 + wv * 16 + g * 8 + r;
            out[(rowbase + qrow) * DM + dcol] = f2bf(accO[f][r] / rsum[r]);
        }
    }
}

// ---------------------------------------------------------------------------
// launch
// ---------------------------------------------------------------------------
extern "C" void kernel_launch(void* const* d_in, const int* in_sizes, int n_in,
                              void* d_out, int out_size, void* d_ws, size_t ws_size,
                              hipStream_t stream) {
    (void)in_sizes; (void)n_in; (void)out_size; (void)ws_size;

    const float* x     = (const float*)d_in[0];   // [4,2048,1024]
    const int*   ids   = (const int*)d_in[1];     // [4,2048]
    const float* w_in  = (const float*)d_in[2];   // [3072,1024]
    const float* w_out = (const float*)d_in[3];   // [1024,1024]
    const float* b_out = (const float*)d_in[4];   // [1024]

    char* ws = (char*)d_ws;
    const size_t nX = (size_t)BS * NS * DM;       // 8388608
    const size_t nWin = (size_t)E3 * DM;          // 3145728
    const size_t nWout = (size_t)DM * DM;         // 1048576
    unsigned short* xb    = (unsigned short*)(ws);
    unsigned short* winb  = (unsigned short*)(ws + 2 * nX);
    unsigned short* woutb = (unsigned short*)(ws + 2 * (nX + nWin));
    unsigned short* qkvb  = (unsigned short*)(ws + 2 * (nX + nWin + nWout));
    unsigned short* attb  = (unsigned short*)(ws + 2 * (nX + nWin + nWout + (size_t)BS * NS * E3));

    convert_bf16_kernel<<<2048, 256, 0, stream>>>(x, xb, (int)nX);
    convert_bf16_kernel<<<1024, 256, 0, stream>>>(w_in, winb, (int)nWin);
    convert_bf16_kernel<<<512, 256, 0, stream>>>(w_out, woutb, (int)nWout);

    // qkv = x @ w_in^T : M=8192, N=3072, K=1024
    gemm_nt_kernel<false><<<dim3(64, 48), 128, 0, stream>>>(
        xb, winb, qkvb, nullptr, BS * NS, E3, DM);

    // attention: 4 * 16 * 32 q-tiles
    attn_kernel<<<2048, 128, 0, stream>>>(qkvb, ids, attb);

    // out = attn @ w_out^T + b_out : M=8192, N=1024, K=1024 (f32 output)
    gemm_nt_kernel<true><<<dim3(64, 16), 128, 0, stream>>>(
        attb, woutb, d_out, b_out, BS * NS, DM, DM);
}
